// GatedAttnLayer_89361089561128
// MI455X (gfx1250) — matrix operand order repaired and checked by reading, not compiled, not run on hardware
//
#include <hip/hip_runtime.h>

#define DEV __device__ __forceinline__

typedef __attribute__((ext_vector_type(16))) __bf16 v16bf;
typedef __attribute__((ext_vector_type(8)))  float  v8f;
typedef __attribute__((ext_vector_type(4)))  unsigned int v4u;
typedef __attribute__((ext_vector_type(8)))  int v8i;
typedef __attribute__((ext_vector_type(4)))  int v4i;

union BF16x16 { v16bf v; unsigned short u[16]; v4u q[2]; };

constexpr int Bc = 8, Nc = 512, Dc = 1024, Hc = 64, DKc = 16;
constexpr int TOK = Bc * Nc; // 4096 tokens

DEV unsigned short f2bf(float f) {
    unsigned u = __builtin_bit_cast(unsigned, f);
    u += 0x7FFFu + ((u >> 16) & 1u);   // round-to-nearest-even
    return (unsigned short)(u >> 16);
}

DEV v8f wmma_bf16(v16bf a, v16bf b, v8f c) {
    return __builtin_amdgcn_wmma_f32_16x16x32_bf16(
        false, a, false, b, (short)0, c, false, false);
}

// ---------------------------------------------------------------- convert (inp)
__global__ void cvt_kernel(const float* __restrict__ src,
                           unsigned short* __restrict__ dst, int n) {
    int i = blockIdx.x * blockDim.x + threadIdx.x;
    if (i < n) dst[i] = f2bf(src[i]);
}

// ------------------------------------------------- transpose-convert (weights)
// dst[col*1024 + k] = bf16(src[k*1024 + col]); 32x32 tiles via LDS.
__global__ void cvtT_kernel(const float* __restrict__ src,
                            unsigned short* __restrict__ dst) {
    __shared__ float tile[32][33];
    int c0 = blockIdx.x * 32, r0 = blockIdx.y * 32;
    for (int i = threadIdx.y; i < 32; i += 8)
        tile[i][threadIdx.x] = src[(size_t)(r0 + i) * Dc + c0 + threadIdx.x];
    __syncthreads();
    for (int i = threadIdx.y; i < 32; i += 8)
        dst[(size_t)(c0 + i) * Dc + r0 + threadIdx.x] = f2bf(tile[threadIdx.x][i]);
}

// ---------------------------------------------------------------- QKV GEMM
// Y = inp @ W + b, W pre-transposed (WT[col][k]). 2x2 register-blocked:
// each wave computes a 32(token) x 32(col) macro-tile = 4 WMMA tiles,
// reusing each A/B fragment twice (2 b128 loads per WMMA).
// Output f32 in [B, H, N, DK] layout per matrix.
__global__ void qkv_gemm_kernel(const unsigned short* __restrict__ Abf,
                                const unsigned short* __restrict__ WTbf,
                                const float* __restrict__ bq,
                                const float* __restrict__ bk,
                                const float* __restrict__ bv,
                                float* __restrict__ qkv) {
    int wid  = blockIdx.x * (blockDim.x >> 5) + (threadIdx.x >> 5);
    int lane = threadIdx.x & 31;
    int half = lane >> 4, lm = lane & 15;
    int mat    = wid >> 12;        // 4096 macro-tiles per matrix (128 x 32)
    int rem    = wid & 4095;
    int ttile2 = rem >> 5;         // 0..127 : 32-token macro-row
    int ctile2 = rem & 31;         // 0..31  : 32-col macro-col
    const unsigned short* WT = WTbf + (size_t)mat * Dc * Dc;
    const float* bias = (mat == 0) ? bq : (mat == 1) ? bk : bv;
    const unsigned short* Arow0 = Abf + (size_t)(ttile2 * 32 + lm) * Dc;
    const unsigned short* Arow1 = Arow0 + (size_t)16 * Dc;
    const unsigned short* Brow0 = WT + (size_t)(ctile2 * 32 + lm) * Dc;
    const unsigned short* Brow1 = Brow0 + (size_t)16 * Dc;
    v8f acc00 = {0.f,0.f,0.f,0.f,0.f,0.f,0.f,0.f};
    v8f acc01 = acc00, acc10 = acc00, acc11 = acc00;
    for (int k0 = 0; k0 < Dc; k0 += 32) {
        BF16x16 a0, a1, b0, b1;
        // A lane layout: e<8 -> K = 8*half+e ; e>=8 -> K = 16+8*half+(e-8)
        a0.q[0] = *(const v4u*)(Arow0 + k0 + (half << 3));
        a0.q[1] = *(const v4u*)(Arow0 + k0 + 16 + (half << 3));
        a1.q[0] = *(const v4u*)(Arow1 + k0 + (half << 3));
        a1.q[1] = *(const v4u*)(Arow1 + k0 + 16 + (half << 3));
        // B lane layout: e<8 -> K = 8*half+e ; e>=8 -> K = 16+8*half+(e-8)
        b0.q[0] = *(const v4u*)(Brow0 + k0 + (half << 3));
        b0.q[1] = *(const v4u*)(Brow0 + k0 + 16 + (half << 3));
        b1.q[0] = *(const v4u*)(Brow1 + k0 + (half << 3));
        b1.q[1] = *(const v4u*)(Brow1 + k0 + 16 + (half << 3));
        if (k0 + 128 < Dc) {
            __builtin_prefetch(Arow0 + k0 + 128, 0, 3);
            __builtin_prefetch(Arow1 + k0 + 128, 0, 3);
            __builtin_prefetch(Brow0 + k0 + 128, 0, 3);
            __builtin_prefetch(Brow1 + k0 + 128, 0, 3);
        }
        acc00 = wmma_bf16(a0.v, b0.v, acc00);
        acc01 = wmma_bf16(a0.v, b1.v, acc01);
        acc10 = wmma_bf16(a1.v, b0.v, acc10);
        acc11 = wmma_bf16(a1.v, b1.v, acc11);
    }
    float* out = qkv + (size_t)mat * TOK * Dc;
#pragma unroll
    for (int ti = 0; ti < 2; ++ti) {
#pragma unroll
        for (int ci = 0; ci < 2; ++ci) {
            v8f acc = (ti == 0) ? ((ci == 0) ? acc00 : acc01)
                                : ((ci == 0) ? acc10 : acc11);
            int hcol = ctile2 * 2 + ci;               // head index
            float bb = bias[hcol * 16 + lm];
#pragma unroll
            for (int r = 0; r < 8; ++r) {
                int m = r + (half << 3);
                int token = ttile2 * 32 + ti * 16 + m;
                int btok = token >> 9, n = token & (Nc - 1);
                out[(((size_t)btok * Hc + hcol) * Nc + n) * DKc + lm] = acc[r] + bb;
            }
        }
    }
}

// ---------------------------------------------------------------- gating
__global__ void gate_kernel(const float* __restrict__ qkv,
                            const float* __restrict__ Wfq, const float* __restrict__ bfq,
                            const float* __restrict__ Wfk, const float* __restrict__ bfk,
                            const float* __restrict__ Wfg, const float* __restrict__ bfg,
                            unsigned short* __restrict__ qg,
                            unsigned short* __restrict__ kg,
                            unsigned short* __restrict__ vb) {
    __shared__ float sWfq[256], sWfk[256], sWfg[512], sbq[16], sbk[16], sbg[32];
    for (int i = threadIdx.x; i < 256; i += blockDim.x) { sWfq[i] = Wfq[i]; sWfk[i] = Wfk[i]; }
    for (int i = threadIdx.x; i < 512; i += blockDim.x) sWfg[i] = Wfg[i];
    if (threadIdx.x < 16) { sbq[threadIdx.x] = bfq[threadIdx.x]; sbk[threadIdx.x] = bfk[threadIdx.x]; }
    if (threadIdx.x < 32) sbg[threadIdx.x] = bfg[threadIdx.x];
    __syncthreads();

    int t = blockIdx.x * blockDim.x + threadIdx.x;   // 0 .. B*H*N-1
    const float4* q4 = (const float4*)(qkv + (size_t)t * DKc);
    const float4* k4 = (const float4*)(qkv + (size_t)TOK * Dc + (size_t)t * DKc);
    const float4* v4 = (const float4*)(qkv + (size_t)2 * TOK * Dc + (size_t)t * DKc);
    float qv[16], kv[16], vv[16], G[16];
#pragma unroll
    for (int i = 0; i < 4; ++i) {
        float4 a = q4[i], b = k4[i], c = v4[i];
        qv[4*i] = a.x; qv[4*i+1] = a.y; qv[4*i+2] = a.z; qv[4*i+3] = a.w;
        kv[4*i] = b.x; kv[4*i+1] = b.y; kv[4*i+2] = b.z; kv[4*i+3] = b.w;
        vv[4*i] = c.x; vv[4*i+1] = c.y; vv[4*i+2] = c.z; vv[4*i+3] = c.w;
    }
#pragma unroll
    for (int i = 0; i < 16; ++i) {
        float s = sbq[i], s2 = sbk[i];
#pragma unroll
        for (int j = 0; j < 16; ++j) { s += qv[j] * sWfq[j * 16 + i]; s2 += kv[j] * sWfk[j * 16 + i]; }
        G[i] = s * s2;
    }
    float Mv[32];
#pragma unroll
    for (int i = 0; i < 32; ++i) {
        float s = sbg[i];
#pragma unroll
        for (int j = 0; j < 16; ++j) s += G[j] * sWfg[j * 32 + i];
        Mv[i] = 1.0f / (1.0f + __expf(-s));
    }
    BF16x16 qo, ko, vo;
#pragma unroll
    for (int i = 0; i < 16; ++i) {
        qo.u[i] = f2bf(qv[i] * Mv[i]);
        ko.u[i] = f2bf(kv[i] * Mv[16 + i]);
        vo.u[i] = f2bf(vv[i]);
    }
    v4u* qd = (v4u*)(qg + (size_t)t * DKc);
    v4u* kd = (v4u*)(kg + (size_t)t * DKc);
    v4u* vd = (v4u*)(vb + (size_t)t * DKc);
    qd[0] = qo.q[0]; qd[1] = qo.q[1];
    kd[0] = ko.q[0]; kd[1] = ko.q[1];
    vd[0] = vo.q[0]; vd[1] = vo.q[1];
}

// ---------------------------------------------------------------- attention
// One workgroup (128 thr) per (b,h). K/V DMA'd to LDS via Tensor Data Mover,
// flash-style online softmax, all WMMA operands via b128 loads.
__global__ void attn_kernel(const unsigned short* __restrict__ qg,
                            const unsigned short* __restrict__ kg,
                            const unsigned short* __restrict__ vb,
                            float* __restrict__ xout) {
    extern __shared__ char smem[];
    unsigned short* sK   = (unsigned short*)smem;            // [512][16] @ 0     (16KB)
    unsigned short* sVs  = (unsigned short*)(smem + 16384);  // [512][16] staging (16KB)
    unsigned short* sVT  = (unsigned short*)(smem + 32768);  // [16][512]         (16KB)
    int bh = blockIdx.x;
    int b = bh >> 6, h = bh & 63;
    const unsigned short* Kg = kg + (size_t)bh * (Nc * DKc);
    const unsigned short* Vg = vb + (size_t)bh * (Nc * DKc);
    const unsigned short* Qg = qg + (size_t)bh * (Nc * DKc);

    // --- TDM: DMA K -> LDS@0 and V -> LDS@16384 (2D tensor, 2B elems, 16 x 512)
    if (threadIdx.x == 0) {
        // D# group1: data_size=2B; dim0=16, dim1=512; tile 16x512; dim0_stride=16
        v8i g1 = { 0x00010000, 0x00100000, 0x02000000, 0x00100000, 512, 16, 0, 0 };
        v4i gz4 = { 0, 0, 0, 0 };
        v8i gz8 = { 0, 0, 0, 0, 0, 0, 0, 0 };
        unsigned long long gaK = (unsigned long long)(uintptr_t)Kg;
        v4u g0K = { 1u, 0u, (unsigned)gaK,
                    ((unsigned)(gaK >> 32) & 0x01FFFFFFu) | 0x80000000u };
        __builtin_amdgcn_tensor_load_to_lds(g0K, g1, gz4, gz4, gz8, 0);
        unsigned long long gaV = (unsigned long long)(uintptr_t)Vg;
        v4u g0V = { 1u, 16384u, (unsigned)gaV,
                    ((unsigned)(gaV >> 32) & 0x01FFFFFFu) | 0x80000000u };
        __builtin_amdgcn_tensor_load_to_lds(g0V, g1, gz4, gz4, gz8, 0);
    }
    __builtin_amdgcn_s_wait_tensorcnt(0);
    __syncthreads();

    // transpose V: sVT[dk][key] = sVs[key][dk]
    for (int i = threadIdx.x; i < Nc * DKc; i += blockDim.x) {
        int key = i >> 4, dk = i & 15;
        sVT[dk * Nc + key] = sVs[i];
    }
    __syncthreads();

    int w = threadIdx.x >> 5, lane = threadIdx.x & 31;
    int half = lane >> 4, lm = lane & 15;
    unsigned short* pt16 = (unsigned short*)(smem + 49152) + w * 512; // 16x32 bf16

    for (int qt = w; qt < 32; qt += 4) {                 // 8 query tiles per wave
        BF16x16 aq;
        // e<8 -> dk = 8*half+e (real); e>=8 -> dk >= 16 (zero pad)
        aq.q[0] = *(const v4u*)(Qg + (qt * 16 + lm) * DKc + (half << 3));
        aq.q[1] = (v4u){0u, 0u, 0u, 0u};
        v8f acc = {0.f, 0.f, 0.f, 0.f, 0.f, 0.f, 0.f, 0.f};
        float mrun[8], lrun[8];
#pragma unroll
        for (int r = 0; r < 8; ++r) { mrun[r] = -1e30f; lrun[r] = 0.f; }

        for (int ktp = 0; ktp < 16; ++ktp) {             // 32 keys per iteration
            // B = K^T : col = key, e<8 -> dk = 8*half+e ; e>=8 -> zero pad
            BF16x16 b0, b1;
            b0.q[0] = b0.q[1] = b1.q[0] = b1.q[1] = (v4u){0u, 0u, 0u, 0u};
            {
                const unsigned short* k0p = sK + (ktp * 32 + lm) * DKc;
                const unsigned short* k1p = sK + (ktp * 32 + 16 + lm) * DKc;
                b0.q[0] = *(const v4u*)(k0p + (half << 3));
                b1.q[0] = *(const v4u*)(k1p + (half << 3));
            }
            v8f z = {0.f, 0.f, 0.f, 0.f, 0.f, 0.f, 0.f, 0.f};
            v8f s0 = wmma_bf16(aq.v, b0.v, z);
            v8f s1 = wmma_bf16(aq.v, b1.v, z);

            // online softmax across this 32-key block
#pragma unroll
            for (int r = 0; r < 8; ++r) {
                float x0 = s0[r] * 0.25f;                // 1/sqrt(16)
                float x1 = s1[r] * 0.25f;
                float a = fmaxf(x0, x1);
                a = fmaxf(a, __shfl_xor(a, 1));
                a = fmaxf(a, __shfl_xor(a, 2));
                a = fmaxf(a, __shfl_xor(a, 4));
                a = fmaxf(a, __shfl_xor(a, 8));
                float mnew = fmaxf(mrun[r], a);
                float p0 = __expf(x0 - mnew);
                float p1 = __expf(x1 - mnew);
                float ps = p0 + p1;
                ps += __shfl_xor(ps, 1);
                ps += __shfl_xor(ps, 2);
                ps += __shfl_xor(ps, 4);
                ps += __shfl_xor(ps, 8);
                float alpha = __expf(mrun[r] - mnew);
                lrun[r] = lrun[r] * alpha + ps;
                mrun[r] = mnew;
                acc[r] *= alpha;
                int m = r + (half << 3);
                pt16[m * 32 + lm]      = f2bf(p0);       // D-layout -> LDS transpose
                pt16[m * 32 + 16 + lm] = f2bf(p1);
            }
            __builtin_amdgcn_wave_barrier();

            // P @ V : A = P 16x32 (LDS, bf16), B = V^T rows (LDS, table order)
            BF16x16 ap, bvv;
            ap.q[0] = *(const v4u*)(pt16 + lm * 32 + (half << 3));
            ap.q[1] = *(const v4u*)(pt16 + lm * 32 + 16 + (half << 3));
            const unsigned short* vp = sVT + lm * Nc + ktp * 32 + (half << 3);
            bvv.q[0] = *(const v4u*)(vp);
            bvv.q[1] = *(const v4u*)(vp + 16);
            acc = wmma_bf16(ap.v, bvv.v, acc);
            __builtin_amdgcn_wave_barrier();
        }
#pragma unroll
        for (int r = 0; r < 8; ++r) {
            int m = r + (half << 3);
            int n = qt * 16 + m;
            xout[((size_t)(b * Nc + n)) * Dc + h * DKc + lm] = acc[r] / lrun[r];
        }
    }
}

// ---------------------------------------------------------------- batchnorm
__global__ void bnstat_kernel(const float* __restrict__ x,
                              float* __restrict__ mean, float* __restrict__ rstd) {
    int f = blockIdx.x * blockDim.x + threadIdx.x;   // feature 0..1023
    float s = 0.f, s2 = 0.f;
    for (int t = 0; t < TOK; ++t) {
        float v = x[(size_t)t * Dc + f];
        s += v; s2 += v * v;
    }
    float mu = s * (1.0f / TOK);
    float var = s2 * (1.0f / TOK) - mu * mu;
    mean[f] = mu;
    rstd[f] = rsqrtf(var + 1e-5f);
}

__global__ void bnapply_kernel(const float* __restrict__ x,
                               const float* __restrict__ mean,
                               const float* __restrict__ rstd,
                               const float* __restrict__ gamma,
                               const float* __restrict__ beta,
                               float* __restrict__ out) {
    size_t i4 = (size_t)blockIdx.x * blockDim.x + threadIdx.x;
    int f = (int)((i4 * 4) & (Dc - 1));
    float4 xv = ((const float4*)x)[i4];
    float4 mv = *(const float4*)(mean + f);
    float4 rv = *(const float4*)(rstd + f);
    float4 gv = *(const float4*)(gamma + f);
    float4 bv = *(const float4*)(beta + f);
    float4 o;
    o.x = (xv.x - mv.x) * rv.x * gv.x + bv.x;
    o.y = (xv.y - mv.y) * rv.y * gv.y + bv.y;
    o.z = (xv.z - mv.z) * rv.z * gv.z + bv.z;
    o.w = (xv.w - mv.w) * rv.w * gv.w + bv.w;
    ((float4*)out)[i4] = o;
}

// ---------------------------------------------------------------- launch
extern "C" void kernel_launch(void* const* d_in, const int* in_sizes, int n_in,
                              void* d_out, int out_size, void* d_ws, size_t ws_size,
                              hipStream_t stream) {
    const float* inp   = (const float*)d_in[0];
    const float* Wq    = (const float*)d_in[1];
    const float* bq    = (const float*)d_in[2];
    const float* Wk    = (const float*)d_in[3];
    const float* bk    = (const float*)d_in[4];
    const float* Wv    = (const float*)d_in[5];
    const float* bv    = (const float*)d_in[6];
    const float* Wfq   = (const float*)d_in[7];
    const float* bfq   = (const float*)d_in[8];
    const float* Wfk   = (const float*)d_in[9];
    const float* bfk   = (const float*)d_in[10];
    const float* Wfg   = (const float*)d_in[11];
    const float* bfg   = (const float*)d_in[12];
    const float* gamma = (const float*)d_in[13];
    const float* beta  = (const float*)d_in[14];
    float* out = (float*)d_out;
    char* ws = (char*)d_ws;

    // workspace layout (bytes)
    unsigned short* inp_bf = (unsigned short*)(ws);               //  8 MB
    unsigned short* WT_bf  = (unsigned short*)(ws + 8388608);     //  6 MB (transposed)
    float*          qkv    = (float*)(ws + 14680064);             // 48 MB [3][B][H][N][DK]
    unsigned short* qgb    = (unsigned short*)(ws + 65011712);    //  8 MB
    unsigned short* kgb    = (unsigned short*)(ws + 73400320);    //  8 MB
    unsigned short* vbf    = (unsigned short*)(ws + 81788928);    //  8 MB
    float*          x      = (float*)(ws + 90177536);             // 16 MB [B*N][D]
    float*          meanp  = (float*)(ws + 106954752);
    float*          rstdp  = (float*)(ws + 106958848);

    cvt_kernel<<<(TOK * Dc) / 256, 256, 0, stream>>>(inp, inp_bf, TOK * Dc);
    dim3 tb(32, 8), tg(32, 32);
    cvtT_kernel<<<tg, tb, 0, stream>>>(Wq, WT_bf);
    cvtT_kernel<<<tg, tb, 0, stream>>>(Wk, WT_bf + Dc * Dc);
    cvtT_kernel<<<tg, tb, 0, stream>>>(Wv, WT_bf + 2 * Dc * Dc);

    // 3 matrices * 128 token-macrotiles * 32 col-macrotiles = 12288 wave-tiles
    qkv_gemm_kernel<<<1536, 256, 0, stream>>>(inp_bf, WT_bf, bq, bk, bv, qkv);

    gate_kernel<<<(Bc * Hc * Nc) / 256, 256, 0, stream>>>(qkv, Wfq, bfq, Wfk, bfk,
                                                          Wfg, bfg, qgb, kgb, vbf);

    attn_kernel<<<Bc * Hc, 128, 53248, stream>>>(qgb, kgb, vbf, x);

    bnstat_kernel<<<Dc / 256, 256, 0, stream>>>(x, meanp, rstdp);
    bnapply_kernel<<<(TOK * Dc / 4) / 256, 256, 0, stream>>>(x, meanp, rstdp,
                                                             gamma, beta, out);
}
